// LGNJSDE_89232240542232
// MI455X (gfx1250) — matrix operand order repaired and checked by reading, not compiled, not run on hardware
//
#include <hip/hip_runtime.h>
#include <hip/hip_bf16.h>
#include <math.h>

// ---------------- problem constants ----------------
#define B_   32
#define S_   32
#define V_   100
#define HD_  32
#define HID_ 64
#define ND_  10
#define VP_  112   // V padded to 7 tiles of 16
#define NT_  7
#define TPB_ 128   // 4 waves of 32 (wave32)
#define VV_  (V_*V_)

// Explicit address spaces so weight streams lower to global_load_* (LOADcnt only)
// and scratch/state to ds_* (DScnt only) instead of FLAT (which ties up both).
#define GAS __attribute__((address_space(1)))
#define LAS __attribute__((address_space(3)))

typedef __attribute__((ext_vector_type(16))) _Float16 v16h;
typedef __attribute__((ext_vector_type(8)))  float    v8f;

// ---------------- d_ws layout (in _Float16 units): transposed, K-padded f16 weights
// Wt[n][k], Kpad multiple of 32, so WMMA B fragments are two 16B runs per lane.
#define OFF_fW1 0        // [64][64]
#define OFF_fW2 4096     // [64][64]
#define OFF_fW3 8192     // [32][64]
#define OFF_gW1 10240    // [64][64]
#define OFF_gW2 14336    // [64][64]
#define OFF_gW3 18432    // [32][64]
#define OFF_eW1 20480    // [64][32]  (N=64, Kpad=32)
#define OFF_eW2 22528    // [16][64]  (N padded 1->16)
#define OFF_mW1 23552    // [64][64]
#define OFF_mW2 27648    // [32][64]
#define WS_HALFS 29696

// ---------------- small helpers ----------------
static __device__ __forceinline__ void waitlds() {
  // CDNA5 split DS counter: ensure all prior LDS ops complete (cross-lane, in-wave)
  asm volatile("s_wait_dscnt 0" ::: "memory");
}

// Branch-free transcendentals on raw CDNA5 VALU trans ops (no exec-mask churn,
// keeps EXEC=all-ones through WMMA regions).
#define LOG2E_ 1.442695040f
#define LN2_   0.6931471806f

static __device__ __forceinline__ float tanh_fast(float x) {
  // tanh(x) = 1 - 2/(e^{2x}+1); exp2 saturates to +inf / 0 -> tanh saturates to +-1
  float e = __builtin_amdgcn_exp2f(x * (2.0f * LOG2E_));
  return 1.0f - 2.0f * __builtin_amdgcn_rcpf(e + 1.0f);
}
static __device__ __forceinline__ float softplus_fast(float x) {
  float e = __builtin_amdgcn_exp2f(x * LOG2E_);
  float l = __builtin_amdgcn_logf(1.0f + e) * LN2_;
  return (x > 20.f) ? x : l;   // cndmask, both sides cheap
}
static __device__ __forceinline__ float sigmoid_fast(float x) {
  // 1/(1+e^{-x})
  float e = __builtin_amdgcn_exp2f(-x * LOG2E_);
  return __builtin_amdgcn_rcpf(1.0f + e);
}

static __device__ __forceinline__ v8f wmma16(v16h a, v16h b, v8f c) {
  // D = A(16x32,f16) * B(32x16,f16) + C(16x16,f32)
  return __builtin_amdgcn_wmma_f32_16x16x32_f16(false, a, false, b, (short)0, c,
                                                false, false);
}

// B fragment (32x16 f16): lane col = lane&15, elements i -> K = (lane>>4)*8 + i,
// elements 8+i -> K = 16 + (lane>>4)*8 + i, offset by kchunk*32 (two 16B-aligned runs).
static __device__ __forceinline__ v16h ldB(const _Float16* Wt, int Kpad, int ntile,
                                           int kchunk, int lane) {
  const GAS _Float16* p = (const GAS _Float16*)(Wt + (size_t)(ntile * 16 + (lane & 15)) * Kpad
                         + kchunk * 32 + ((lane >> 4) * 8));
  v16h b;
#pragma unroll
  for (int i = 0; i < 8; ++i) { b[i] = p[i]; b[8 + i] = p[16 + i]; }
  return b;
}

static __device__ __forceinline__ float ldg(const float* p, int i) {
  return ((const GAS float*)p)[i];
}

// A fragment (16x32 f16) covering K=0..31 from a per-lane f32 LDS row pointer.
// element j   -> K = half*8 + j
// element 8+j -> K = 16 + half*8 + j
static __device__ __forceinline__ v16h A_f32row(const float* row, int lane) {
  const LAS float* r = (const LAS float*)row;
  int hb = (lane >> 4) * 8;
  v16h a;
#pragma unroll
  for (int i = 0; i < 8; ++i) {
    a[i]     = (_Float16)r[hb + i];
    a[8 + i] = (_Float16)r[16 + hb + i];
  }
  return a;
}

// A fragment for the padded K=32..63 chunk where only features 32 (e0) and 33 (e1)
// are real; weight rows >= true K are zero so the rest can be 0.
static __device__ __forceinline__ v16h A_scal2(float e0, float e1, int lane) {
  v16h a;
#pragma unroll
  for (int i = 0; i < 16; ++i) a[i] = (_Float16)0.f;
  if ((lane >> 4) == 0) { a[0] = (_Float16)e0; a[1] = (_Float16)e1; }
  return a;
}

// A fragment from per-wave f16 LDS scratch tile scr[16][64], K chunk at kbase.
static __device__ __forceinline__ v16h A_scr(const _Float16* scr, int kbase, int lane) {
  const LAS _Float16* p =
      (const LAS _Float16*)(scr + (lane & 15) * 64 + kbase + (lane >> 4) * 8);
  v16h a;
#pragma unroll
  for (int i = 0; i < 8; ++i) { a[i] = p[i]; a[8 + i] = p[16 + i]; }
  return a;
}

// One MLP layer: Y = tanh(X@W + b), result written as f16 tile to LDS scratch.
template <int NTILES, int NF>
static __device__ __forceinline__ void layer_to_scr(const v16h (&af)[NF],
                                                    const _Float16* Wt, int Kpad,
                                                    const float* bias,
                                                    _Float16* scr, int lane) {
  int half = lane >> 4, colL = lane & 15;
  LAS _Float16* sp = (LAS _Float16*)scr;
#pragma unroll
  for (int n = 0; n < NTILES; ++n) {
    v8f acc;
#pragma unroll
    for (int r = 0; r < 8; ++r) acc[r] = 0.f;
#pragma unroll
    for (int f = 0; f < NF; ++f) acc = wmma16(af[f], ldB(Wt, Kpad, n, f, lane), acc);
    int col = n * 16 + colL;
    float bb = ldg(bias, col);
#pragma unroll
    for (int r = 0; r < 8; ++r) {
      int m = r + 8 * half;
      sp[m * 64 + col] = (_Float16)tanh_fast(acc[r] + bb);   // ds_store_b16
    }
  }
}

// Final linear layer: D = X@W + b kept in f32 fragments.
template <int NTILES, int NF>
static __device__ __forceinline__ void layer_to_D(const v16h (&af)[NF],
                                                  const _Float16* Wt, int Kpad,
                                                  const float* bias,
                                                  v8f (&Dout)[NTILES], int lane) {
  int colL = lane & 15;
#pragma unroll
  for (int n = 0; n < NTILES; ++n) {
    v8f acc;
#pragma unroll
    for (int r = 0; r < 8; ++r) acc[r] = 0.f;
#pragma unroll
    for (int f = 0; f < NF; ++f) acc = wmma16(af[f], ldB(Wt, Kpad, n, f, lane), acc);
    float bb = ldg(bias, n * 16 + colL);
#pragma unroll
    for (int r = 0; r < 8; ++r) acc[r] += bb;
    Dout[n] = acc;
  }
}

// e_int for one 16-row tile: l = softplus(tanh(h@eW1+eb1)@eW2+eb2)
static __device__ __forceinline__ void e_int_tile(const float* hS, int t,
                                                  const _Float16* WeW1,
                                                  const _Float16* WeW2,
                                                  const float* eb1, float eb2_0,
                                                  float* lout, _Float16* scr,
                                                  int lane) {
  const float* hrow = hS + (size_t)(t * 16 + (lane & 15)) * HD_;
  v16h a1[1];
  a1[0] = A_f32row(hrow, lane);
  layer_to_scr<4>(a1, WeW1, 32, eb1, scr, lane);
  waitlds();
  v16h a2[2];
  a2[0] = A_scr(scr, 0, lane);
  a2[1] = A_scr(scr, 32, lane);
  v8f acc;
#pragma unroll
  for (int r = 0; r < 8; ++r) acc[r] = 0.f;
  acc = wmma16(a2[0], ldB(WeW2, 64, 0, 0, lane), acc);
  acc = wmma16(a2[1], ldB(WeW2, 64, 0, 1, lane), acc);
  if ((lane & 15) == 0) {
    int half = lane >> 4;
    LAS float* lo = (LAS float*)lout;
#pragma unroll
    for (int r = 0; r < 8; ++r) {
      int v = t * 16 + r + 8 * half;
      lo[v] = softplus_fast(acc[r] + eb2_0);
    }
  }
  waitlds();
}

// hash RNG -> approx N(0,1) (exact jax PRNG reproduction is not required here)
static __device__ __forceinline__ unsigned hashu(unsigned x) {
  x ^= x >> 17; x *= 0xed5ad4bbu; x ^= x >> 11; x *= 0xac4c1b51u;
  x ^= x >> 15; x *= 0x31848babu; x ^= x >> 14; return x;
}
static __device__ __forceinline__ float gaussf(unsigned s) {
  unsigned a = hashu(s ^ 0x9E3779B9u);
  unsigned b = hashu(s + 0x7F4A7C15u);
  float u1 = ((a >> 8) + 1u) * (1.f / 16777218.f);
  float u2 = (b >> 8) * (1.f / 16777216.f);
  float lg = __builtin_amdgcn_logf(u1) * LN2_;      // ln(u1)
  return __builtin_amdgcn_sqrtf(-2.f * lg) * __builtin_amdgcn_cosf(u2);
}

// msg_jump: h += tanh(concat(h[ev],h)@mW1+mb1)@mW2+mb2 * p ; then h[ev] += jump(h[ev])
static __device__ void msg_jump(float* hS, _Float16* scr,
                                const _Float16* WmW1, const _Float16* WmW2,
                                const float* mb1, const float* mb2,
                                const float* lg,
                                const float* jW1, const float* jb1,
                                const float* jW2, const float* jb2,
                                float* sendbuf, float* hebuf, float* jtmp,
                                int ev, int tid, int lane, int wave) {
  LAS float* hL = (LAS float*)hS;
  LAS float* sendL = (LAS float*)sendbuf;
  LAS float* heL = (LAS float*)hebuf;
  LAS float* jtL = (LAS float*)jtmp;
  if (tid < HD_) sendL[tid] = hL[ev * HD_ + tid];
  __syncthreads();
  for (int t = wave; t < NT_; t += 4) {
    const float* hrow = hS + (size_t)(t * 16 + (lane & 15)) * HD_;
    v16h af[2];
    af[0] = A_f32row(sendbuf, lane);   // cols 0..31  = h[ev] broadcast
    af[1] = A_f32row(hrow, lane);      // cols 32..63 = h row
    layer_to_scr<4>(af, WmW1, 64, mb1, scr, lane);
    waitlds();
    v16h a2[2];
    a2[0] = A_scr(scr, 0, lane);
    a2[1] = A_scr(scr, 32, lane);
    v8f mD[2];
    layer_to_D<2>(a2, WmW2, 64, mb2, mD, lane);
    int half = lane >> 4, cl = lane & 15;
#pragma unroll
    for (int n = 0; n < 2; ++n) {
#pragma unroll
      for (int r = 0; r < 8; ++r) {
        int v = t * 16 + r + 8 * half;
        int d = n * 16 + cl;
        float p = 0.f;
        if (v < V_) {
          int idx = ev * V_ + v;
          p = sigmoid_fast(ldg(lg, VV_ + idx) - ldg(lg, idx));  // softmax axis0 row 1
        }
        hL[v * HD_ + d] += mD[n][r] * p;
      }
    }
    waitlds();
  }
  __syncthreads();
  if (tid < HD_) heL[tid] = hL[ev * HD_ + tid];
  __syncthreads();
  if (tid < HD_) {
#pragma unroll
    for (int half = 0; half < 2; ++half) {
      int n = tid + half * 32;
      float acc = ldg(jb1, n);
      for (int k = 0; k < HD_; ++k) acc += heL[k] * ldg(jW1, k * HID_ + n);
      jtL[n] = tanh_fast(acc);
    }
  }
  __syncthreads();
  if (tid < HD_) {
    float acc = ldg(jb2, tid);
    for (int k = 0; k < HID_; ++k) acc += jtL[k] * ldg(jW2, k * HD_ + tid);
    hL[ev * HD_ + tid] += acc;
  }
  __syncthreads();
}

// ---------------- prep: transpose+convert weights to f16 in d_ws, zero nll ----------------
static __device__ void transpose_fill(_Float16* dst, const float* W, int Ktrue, int N,
                                      int Kpad, int Npad, int tid, int nth) {
  int tot = Npad * Kpad;
  for (int i = tid; i < tot; i += nth) {
    int n = i / Kpad, k = i % Kpad;
    float v = (k < Ktrue && n < N) ? W[k * N + n] : 0.f;
    dst[n * Kpad + k] = (_Float16)v;
  }
}

__global__ void lgnjsde_prep(const float* fW1, const float* fW2, const float* fW3,
                             const float* gW1, const float* gW2, const float* gW3,
                             const float* eW1, const float* eW2,
                             const float* mW1, const float* mW2,
                             _Float16* wsh, float* out) {
  int tid = blockIdx.x * blockDim.x + threadIdx.x;
  int nth = gridDim.x * blockDim.x;
  if (tid == 0) out[0] = 0.f;
  transpose_fill(wsh + OFF_fW1, fW1, 34, 64, 64, 64, tid, nth);
  transpose_fill(wsh + OFF_fW2, fW2, 64, 64, 64, 64, tid, nth);
  transpose_fill(wsh + OFF_fW3, fW3, 64, 32, 64, 32, tid, nth);
  transpose_fill(wsh + OFF_gW1, gW1, 33, 64, 64, 64, tid, nth);
  transpose_fill(wsh + OFF_gW2, gW2, 64, 64, 64, 64, tid, nth);
  transpose_fill(wsh + OFF_gW3, gW3, 64, 32, 64, 32, tid, nth);
  transpose_fill(wsh + OFF_eW1, eW1, 32, 64, 32, 64, tid, nth);
  transpose_fill(wsh + OFF_eW2, eW2, 64, 1, 64, 16, tid, nth);
  transpose_fill(wsh + OFF_mW1, mW1, 64, 64, 64, 64, tid, nth);
  transpose_fill(wsh + OFF_mW2, mW2, 64, 32, 64, 32, tid, nth);
}

// ---------------- main: one workgroup per batch element, whole scan on-chip ----------------
__global__ __launch_bounds__(TPB_) void lgnjsde_main(
    const float* __restrict__ bt_time, const int* __restrict__ bt_type,
    const float* __restrict__ logits, const float* __restrict__ h0,
    const float* fb1, const float* fb2, const float* fb3,
    const float* gb1, const float* gb2, const float* gb3,
    const float* eb1, const float* eb2,
    const float* jW1, const float* jb1, const float* jW2, const float* jb2,
    const float* mb1, const float* mb2,
    const _Float16* __restrict__ wsh, float* __restrict__ out) {
  __shared__ float    hS[VP_ * HD_];        // 14.0 KB  state, resident all kernel
  __shared__ _Float16 scrS[4 * 16 * 64];    //  8.0 KB  per-wave activation scratch
  __shared__ float    lb[2][VP_];           //  0.9 KB  intensity double buffer
  __shared__ float    sendbuf[HD_], hebuf[HD_], jtmp[HID_];
  __shared__ float    s_sum[2];             // [0]=sum_term, [1]=integral

  int b = blockIdx.x;
  int tid = threadIdx.x;
  int lane = tid & 31, wave = tid >> 5;
  _Float16* scr = scrS + wave * 16 * 64;
  LAS float* hL = (LAS float*)hS;

  const _Float16* WfW1 = wsh + OFF_fW1;
  const _Float16* WfW2 = wsh + OFF_fW2;
  const _Float16* WfW3 = wsh + OFF_fW3;
  const _Float16* WgW1 = wsh + OFF_gW1;
  const _Float16* WgW2 = wsh + OFF_gW2;
  const _Float16* WgW3 = wsh + OFF_gW3;
  const _Float16* WeW1 = wsh + OFF_eW1;
  const _Float16* WeW2 = wsh + OFF_eW2;
  const _Float16* WmW1 = wsh + OFF_mW1;
  const _Float16* WmW2 = wsh + OFF_mW2;
  float* outL = out + 1;

  __builtin_prefetch(wsh, 0, 1);  // global_prefetch: pull weights toward WGP

  // init h (broadcast h0, zero the 12 pad rows), zero accumulators
  for (int i = tid; i < VP_ * HD_; i += TPB_) {
    int v = i / HD_;
    hL[i] = (v < V_) ? ldg(h0, i) : 0.f;
  }
  if (tid < 2) s_sum[tid] = 0.f;
  __syncthreads();

  float eb2_0 = ldg(eb2, 0);

  // l0 (before first msg_jump), lt0
  for (int t = wave; t < NT_; t += 4)
    e_int_tile(hS, t, WeW1, WeW2, eb1, eb2_0, lb[0], scr, lane);
  __syncthreads();
  int ev0 = bt_type[b * S_ + 0];
  for (int v = tid; v < V_; v += TPB_) outL[(size_t)(b * S_ + 0) * V_ + v] = lb[0][v];
  if (tid == 0) s_sum[0] += __builtin_amdgcn_logf(lb[0][ev0] + 1e-16f) * LN2_;
  __syncthreads();
  msg_jump(hS, scr, WmW1, WmW2, mb1, mb2, logits, jW1, jb1, jW2, jb2,
           sendbuf, hebuf, jtmp, ev0, tid, lane, wave);

  float integ = 0.f;
  for (int s = 1; s < S_; ++s) {
    float t0 = bt_time[b * S_ + s - 1];
    float t1 = bt_time[b * S_ + s];
    int ev = bt_type[b * S_ + s];
    float dt = (t1 - t0) / (float)ND_;
    float sq = __builtin_amdgcn_sqrtf(dt);

    // l_init for this interval
    for (int t = wave; t < NT_; t += 4)
      e_int_tile(hS, t, WeW1, WeW2, eb1, eb2_0, lb[0], scr, lane);
    __syncthreads();

    for (int j = 0; j < ND_; ++j) {
      float dft = dt * (float)(j + 1);
      for (int t = wave; t < NT_; t += 4) {
        const float* hrow = hS + (size_t)(t * 16 + (lane & 15)) * HD_;
        v16h af[2];
        af[0] = A_f32row(hrow, lane);        // h features (K 0..31)
        af[1] = A_scal2(dft, t0, lane);      // [dft, last] at K=32,33
        // f_drift: 34->64->64->32
        layer_to_scr<4>(af, WfW1, 64, fb1, scr, lane);
        waitlds();
        v16h a2[2];
        a2[0] = A_scr(scr, 0, lane); a2[1] = A_scr(scr, 32, lane);
        layer_to_scr<4>(a2, WfW2, 64, fb2, scr, lane);
        waitlds();
        a2[0] = A_scr(scr, 0, lane); a2[1] = A_scr(scr, 32, lane);
        v8f fD[2];
        layer_to_D<2>(a2, WfW3, 64, fb3, fD, lane);
        // g_diff: 33->64->64->32 (reuses af[0] = old h, still in regs)
        af[1] = A_scal2(dft, 0.f, lane);     // pad col 33 has zero weights
        layer_to_scr<4>(af, WgW1, 64, gb1, scr, lane);
        waitlds();
        a2[0] = A_scr(scr, 0, lane); a2[1] = A_scr(scr, 32, lane);
        layer_to_scr<4>(a2, WgW2, 64, gb2, scr, lane);
        waitlds();
        a2[0] = A_scr(scr, 0, lane); a2[1] = A_scr(scr, 32, lane);
        v8f gD[2];
        layer_to_D<2>(a2, WgW3, 64, gb3, gD, lane);
        // Euler-Maruyama update (each wave owns its tile rows: no cross-wave sync)
        int half = lane >> 4, cl = lane & 15;
#pragma unroll
        for (int n = 0; n < 2; ++n) {
#pragma unroll
          for (int r = 0; r < 8; ++r) {
            int v = t * 16 + r + 8 * half;
            int d = n * 16 + cl;
            unsigned sd = ((unsigned)(b * S_ + s) * ND_ + (unsigned)j) * 4096u
                        + (unsigned)(v * HD_ + d);
            float nz = gaussf(sd);
            hL[v * HD_ + d] += fD[n][r] * dt + gD[n][r] * sq * nz;
          }
        }
        waitlds();
        e_int_tile(hS, t, WeW1, WeW2, eb1, eb2_0, lb[(j + 1) & 1], scr, lane);
      }
      __syncthreads();
      const float* lp = lb[j & 1];
      const float* lc = lb[(j + 1) & 1];
      for (int v = tid; v < V_; v += TPB_) integ += 0.5f * dt * (lp[v] + lc[v]);
      __syncthreads();
    }
    // ND even -> l_last lives in lb[0]
    for (int v = tid; v < V_; v += TPB_) outL[(size_t)(b * S_ + s) * V_ + v] = lb[0][v];
    if (tid == 0) s_sum[0] += __builtin_amdgcn_logf(lb[0][ev] + 1e-16f) * LN2_;
    __syncthreads();
    msg_jump(hS, scr, WmW1, WmW2, mb1, mb2, logits, jW1, jb1, jW2, jb2,
             sendbuf, hebuf, jtmp, ev, tid, lane, wave);
  }

  atomicAdd(&s_sum[1], integ);
  __syncthreads();
  if (tid == 0) atomicAdd(out, s_sum[1] - s_sum[0]);
}

// ---------------- launcher ----------------
extern "C" void kernel_launch(void* const* d_in, const int* in_sizes, int n_in,
                              void* d_out, int out_size, void* d_ws, size_t ws_size,
                              hipStream_t stream) {
  (void)in_sizes; (void)n_in; (void)out_size; (void)ws_size;
  const float* bt_time = (const float*)d_in[0];
  const int*   bt_type = (const int*)d_in[1];
  // d_in[2] = mask (all ones) unused
  const float* logits = (const float*)d_in[3];
  const float* h0     = (const float*)d_in[4];
  const float* fW1 = (const float*)d_in[5];  const float* fb1 = (const float*)d_in[6];
  const float* fW2 = (const float*)d_in[7];  const float* fb2 = (const float*)d_in[8];
  const float* fW3 = (const float*)d_in[9];  const float* fb3 = (const float*)d_in[10];
  const float* gW1 = (const float*)d_in[11]; const float* gb1 = (const float*)d_in[12];
  const float* gW2 = (const float*)d_in[13]; const float* gb2 = (const float*)d_in[14];
  const float* gW3 = (const float*)d_in[15]; const float* gb3 = (const float*)d_in[16];
  const float* eW1 = (const float*)d_in[17]; const float* eb1 = (const float*)d_in[18];
  const float* eW2 = (const float*)d_in[19]; const float* eb2 = (const float*)d_in[20];
  const float* jW1 = (const float*)d_in[21]; const float* jb1 = (const float*)d_in[22];
  const float* jW2 = (const float*)d_in[23]; const float* jb2 = (const float*)d_in[24];
  const float* mW1 = (const float*)d_in[25]; const float* mb1 = (const float*)d_in[26];
  const float* mW2 = (const float*)d_in[27]; const float* mb2 = (const float*)d_in[28];
  _Float16* wsh = (_Float16*)d_ws;
  float* out = (float*)d_out;

  lgnjsde_prep<<<dim3(1), dim3(256), 0, stream>>>(fW1, fW2, fW3, gW1, gW2, gW3,
                                                  eW1, eW2, mW1, mW2, wsh, out);
  lgnjsde_main<<<dim3(B_), dim3(TPB_), 0, stream>>>(
      bt_time, bt_type, logits, h0, fb1, fb2, fb3, gb1, gb2, gb3, eb1, eb2,
      jW1, jb1, jW2, jb2, mb1, mb2, (const _Float16*)wsh, out);
}
